// MolecularNetwork_36077725286789
// MI455X (gfx1250) — matrix operand order repaired and checked by reading, not compiled, not run on hardware
//
#include <hip/hip_runtime.h>

typedef __attribute__((ext_vector_type(16))) _Float16 v16h;
typedef __attribute__((ext_vector_type(8)))  _Float16 v8h;
typedef __attribute__((ext_vector_type(8)))  float    v8f;

#define HID   128
#define HEADS 8
#define CDIM  16
#define LAYERS 3
#define NGRAPH 500
#define NEG_SLOPE 0.2f
#define LN_EPS 1e-5f

// ---------------------------------------------------------------------------
// helpers
// ---------------------------------------------------------------------------
__device__ inline void atomicMaxF(float* addr, float val) {
    int* ia = (int*)addr;
    int cur = *ia;
    while (__int_as_float(cur) < val) {
        int prev = atomicCAS(ia, cur, __float_as_int(val));
        if (prev == cur) break;
        cur = prev;
    }
}

__global__ void fill_kernel(float* __restrict__ p, float v, int n) {
    int i = blockIdx.x * blockDim.x + threadIdx.x;
    if (i < n) p[i] = v;
}

// convert + transpose weights to f16, N-major (Bt[n][k])
__global__ void prep_weights_kernel(const float* __restrict__ embW,
                                    const float* __restrict__ gatW,
                                    const float* __restrict__ gaW1,
                                    _Float16* __restrict__ embWt,
                                    _Float16* __restrict__ gatWt,
                                    _Float16* __restrict__ gaW1t) {
    int i = blockIdx.x * blockDim.x + threadIdx.x;
    if (i < HID * 64) {                       // emb_W: [64][128] -> [128][64]
        int n = i / 64, k = i % 64;
        embWt[i] = (_Float16)embW[k * HID + n];
    }
    if (i < LAYERS * HID * HID) {             // gat_W[l]: [128][128] -> [128][128]^T
        int l = i / (HID * HID);
        int r = i % (HID * HID);
        int n = r / HID, k = r % HID;
        gatWt[i] = (_Float16)gatW[(size_t)l * HID * HID + k * HID + n];
    }
    if (i < HID * HID) {                      // ga_W1
        int n = i / HID, k = i % HID;
        gaW1t[i] = (_Float16)gaW1[k * HID + n];
    }
}

// ---------------------------------------------------------------------------
// WMMA GEMM: C[M][128] = A[M][K] (f32, cvt->f16) @ B[K][128] (pre-transposed f16)
// block = 8 waves; wave w -> rows [blk*128 + w*16, +16), all 128 cols.
// Out-of-range rows read clamped row 0 (harmless) and are masked at store.
// ---------------------------------------------------------------------------
__global__ __launch_bounds__(256)
void gemm_wmma_kernel(const float* __restrict__ A,
                      const _Float16* __restrict__ Bt,
                      const float* __restrict__ bias,         // [128] or null
                      const float* __restrict__ ntype_emb,    // [10][128] or null
                      const int*   __restrict__ node_types,   // [M] or null
                      float* __restrict__ Cout, int M, int K) {
    const int lane = threadIdx.x & 31;
    const int wave = threadIdx.x >> 5;
    const int rowBase = blockIdx.x * 128 + wave * 16;
    const int half = lane >> 4;     // 0 | 1
    const int l16  = lane & 15;

    v8f acc[8] = {};

    const int arow = rowBase + l16;
    const float* Arow = A + (size_t)(arow < M ? arow : 0) * K;   // clamp: stores are guarded

    for (int kk = 0; kk < K; kk += 32) {
        // A fragment (16-bit A layout): lane holds K chunks [b,b+8) and [16+b,16+b+8), b=8*half
        const float* p0 = Arow + kk + half * 8;
        const float* p1 = Arow + kk + 16 + half * 8;
        float4 x0 = *(const float4*)(p0);
        float4 x1 = *(const float4*)(p0 + 4);
        float4 x2 = *(const float4*)(p1);
        float4 x3 = *(const float4*)(p1 + 4);
        v16h a;
        a[0]  = (_Float16)x0.x; a[1]  = (_Float16)x0.y; a[2]  = (_Float16)x0.z; a[3]  = (_Float16)x0.w;
        a[4]  = (_Float16)x1.x; a[5]  = (_Float16)x1.y; a[6]  = (_Float16)x1.z; a[7]  = (_Float16)x1.w;
        a[8]  = (_Float16)x2.x; a[9]  = (_Float16)x2.y; a[10] = (_Float16)x2.z; a[11] = (_Float16)x2.w;
        a[12] = (_Float16)x3.x; a[13] = (_Float16)x3.y; a[14] = (_Float16)x3.z; a[15] = (_Float16)x3.w;

#pragma unroll
        for (int ct = 0; ct < 8; ++ct) {
            // B fragment: lane holds col = ct*16 + l16, K chunks [b,b+8) and [16+b,16+b+8), b=8*half
            const _Float16* bp = Bt + (size_t)(ct * 16 + l16) * K + kk + half * 8;
            v8h b0 = *(const v8h*)(bp);
            v8h b1 = *(const v8h*)(bp + 16);
            v16h b = __builtin_shufflevector(b0, b1,
                        0, 1, 2, 3, 4, 5, 6, 7, 8, 9, 10, 11, 12, 13, 14, 15);
            acc[ct] = __builtin_amdgcn_wmma_f32_16x16x32_f16(
                false, a, false, b, (short)0, acc[ct], false, false);
        }
    }

    // epilogue: C layout -> VGPR v is row (v + 8*half), col = l16
#pragma unroll
    for (int ct = 0; ct < 8; ++ct) {
        const int col = ct * 16 + l16;
#pragma unroll
        for (int v = 0; v < 8; ++v) {
            const int row = rowBase + v + 8 * half;
            if (row < M) {
                float val = acc[ct][v];
                if (bias) val += bias[col];
                if (ntype_emb) val += ntype_emb[node_types[row] * HID + col];
                Cout[(size_t)row * HID + col] = val;
            }
        }
    }
}

// ---------------------------------------------------------------------------
// per-(node, head) attention logits: alpha_s/d[n][h] = sum_c hp[n][h*16+c]*att[h][c]
// ---------------------------------------------------------------------------
__global__ void alpha_kernel(const float* __restrict__ hp,
                             const float* __restrict__ att_src,
                             const float* __restrict__ att_dst,
                             float* __restrict__ as_, float* __restrict__ ad_, int N_) {
    int i = blockIdx.x * blockDim.x + threadIdx.x;
    if (i >= N_ * HEADS) return;
    int h = i & (HEADS - 1), n = i >> 3;
    const float* row = hp + (size_t)n * HID + h * CDIM;
    const float* s = att_src + h * CDIM;
    const float* d = att_dst + h * CDIM;
    float a = 0.f, b = 0.f;
#pragma unroll
    for (int c = 0; c < CDIM; ++c) { float x = row[c]; a += x * s[c]; b += x * d[c]; }
    as_[i] = a; ad_[i] = b;
}

__device__ inline void edge_sd(const int* ei, int e, int E_, int& s, int& d) {
    if (e < E_) { s = ei[e]; d = ei[E_ + e]; }
    else        { s = e - E_; d = e - E_; }      // appended self loops
}

__global__ void edge_max_kernel(const int* __restrict__ ei,
                                const float* __restrict__ as_, const float* __restrict__ ad_,
                                float* __restrict__ smax, int E_, int Etot) {
    int i = blockIdx.x * blockDim.x + threadIdx.x;
    if (i >= Etot * HEADS) return;
    int h = i & (HEADS - 1), e = i >> 3, s, d;
    edge_sd(ei, e, E_, s, d);
    float v = as_[s * HEADS + h] + ad_[d * HEADS + h];
    v = (v >= 0.f) ? v : NEG_SLOPE * v;
    atomicMaxF(&smax[d * HEADS + h], v);
}

__global__ void edge_sum_kernel(const int* __restrict__ ei,
                                const float* __restrict__ as_, const float* __restrict__ ad_,
                                const float* __restrict__ smax, float* __restrict__ ssum,
                                int E_, int Etot) {
    int i = blockIdx.x * blockDim.x + threadIdx.x;
    if (i >= Etot * HEADS) return;
    int h = i & (HEADS - 1), e = i >> 3, s, d;
    edge_sd(ei, e, E_, s, d);
    float v = as_[s * HEADS + h] + ad_[d * HEADS + h];
    v = (v >= 0.f) ? v : NEG_SLOPE * v;
    atomicAdd(&ssum[d * HEADS + h], __expf(v - smax[d * HEADS + h]));
}

// thread per (edge, feature): the bandwidth-dominant pass
__global__ void edge_agg_kernel(const int* __restrict__ ei,
                                const float* __restrict__ as_, const float* __restrict__ ad_,
                                const float* __restrict__ smax, const float* __restrict__ ssum,
                                const float* __restrict__ hp, float* __restrict__ agg,
                                int E_, int Etot) {
    long long i = (long long)blockIdx.x * blockDim.x + threadIdx.x;
    if (i >= (long long)Etot * HID) return;
    int f = (int)(i & (HID - 1));
    int e = (int)(i >> 7);
    int h = f >> 4, s, d;
    edge_sd(ei, e, E_, s, d);
    float v = as_[s * HEADS + h] + ad_[d * HEADS + h];
    v = (v >= 0.f) ? v : NEG_SLOPE * v;
    float alpha = __expf(v - smax[d * HEADS + h]) / ssum[d * HEADS + h];
    atomicAdd(&agg[(size_t)d * HID + f], hp[(size_t)s * HID + f] * alpha);
}

// ---------------------------------------------------------------------------
// one wave32 per node: h = relu(layernorm(agg + gat_b)*g + b + h_prev)
// ---------------------------------------------------------------------------
__global__ void ln_res_kernel(const float* __restrict__ agg, const float* __restrict__ gb,
                              const float* __restrict__ g, const float* __restrict__ b,
                              float* __restrict__ h, int N_) {
    int node = (blockIdx.x * blockDim.x + threadIdx.x) >> 5;
    int lane = threadIdx.x & 31;
    if (node >= N_) return;
    const float* arow = agg + (size_t)node * HID;
    float* hrow = h + (size_t)node * HID;
    float vals[4], hprev[4], sum = 0.f;
#pragma unroll
    for (int j = 0; j < 4; ++j) {
        int f = lane * 4 + j;
        vals[j] = arow[f] + gb[f];
        hprev[j] = hrow[f];
        sum += vals[j];
    }
    for (int o = 16; o > 0; o >>= 1) sum += __shfl_xor(sum, o, 32);
    float mu = sum * (1.f / HID), vs = 0.f;
#pragma unroll
    for (int j = 0; j < 4; ++j) { float dd = vals[j] - mu; vs += dd * dd; }
    for (int o = 16; o > 0; o >>= 1) vs += __shfl_xor(vs, o, 32);
    float r = rsqrtf(vs * (1.f / HID) + LN_EPS);
#pragma unroll
    for (int j = 0; j < 4; ++j) {
        int f = lane * 4 + j;
        float o_ = (vals[j] - mu) * r * g[f] + b[f] + hprev[j];
        hrow[f] = fmaxf(o_, 0.f);
    }
}

// ---------------------------------------------------------------------------
// global attention pooling
// ---------------------------------------------------------------------------
__global__ void score_kernel(const float* __restrict__ hp, const float* __restrict__ gaW2,
                             const float* __restrict__ gab2, float* __restrict__ scores,
                             float* __restrict__ maxsc, int N_) {
    int node = (blockIdx.x * blockDim.x + threadIdx.x) >> 5;
    int lane = threadIdx.x & 31;
    if (node >= N_) return;
    const float* row = hp + (size_t)node * HID;
    float acc = 0.f;
#pragma unroll
    for (int j = 0; j < 4; ++j) { int f = lane * 4 + j; acc += tanhf(row[f]) * gaW2[f]; }
    for (int o = 16; o > 0; o >>= 1) acc += __shfl_xor(acc, o, 32);
    if (lane == 0) {
        float sc = acc + gab2[0];
        scores[node] = sc;
        atomicMaxF(maxsc, sc);
    }
}

__global__ void score_exp_kernel(float* __restrict__ scores, const float* __restrict__ scal,
                                 float* __restrict__ sumsc, int N_) {
    __shared__ float red[256];
    int i = blockIdx.x * 256 + threadIdx.x;
    float w = 0.f;
    if (i < N_) { w = __expf(scores[i] - scal[0]); scores[i] = w; }
    red[threadIdx.x] = w; __syncthreads();
    for (int s = 128; s > 0; s >>= 1) {
        if (threadIdx.x < s) red[threadIdx.x] += red[threadIdx.x + s];
        __syncthreads();
    }
    if (threadIdx.x == 0) atomicAdd(sumsc, red[0]);
}

// 128 blocks x 128 threads: wsum[f] = sum_n h[n][f]*w[n]
__global__ void wsum_kernel(const float* __restrict__ h, const float* __restrict__ scores,
                            float* __restrict__ wsum, int N_) {
    int f = threadIdx.x;
    float acc = 0.f;
    for (int n = blockIdx.x; n < N_; n += gridDim.x)
        acc += h[(size_t)n * HID + f] * scores[n];
    atomicAdd(&wsum[f], acc);
}

__global__ void pool_kernel(const float* __restrict__ h, const int* __restrict__ batch,
                            float* __restrict__ sums, float* __restrict__ cnts, int N_) {
    int i = blockIdx.x * blockDim.x + threadIdx.x;
    if (i >= N_ * HID) return;
    int f = i & (HID - 1), n = i >> 7;
    int g = batch[n];
    atomicAdd(&sums[g * HID + f], h[i]);
    if (f == 0) atomicAdd(&cnts[g], 1.f);
}

// one 128-thread block per graph
__global__ void head_kernel(const float* __restrict__ wsum, const float* __restrict__ scal,
                            const float* __restrict__ sums, const float* __restrict__ cnts,
                            const float* __restrict__ W1, const float* __restrict__ b1,
                            const float* __restrict__ W2, const float* __restrict__ b2,
                            float* __restrict__ out) {
    __shared__ float hg[HID];
    __shared__ float red[HID];
    int g = blockIdx.x, t = threadIdx.x;
    float cnt = fmaxf(cnts[g], 1.f);
    hg[t] = wsum[t] / scal[1] + sums[g * HID + t] / cnt;
    __syncthreads();
    float acc = 0.f;
    for (int j = 0; j < HID; ++j) acc += hg[j] * W1[j * HID + t];
    red[t] = fmaxf(acc + b1[t], 0.f) * W2[t];
    __syncthreads();
    for (int s = 64; s > 0; s >>= 1) {
        if (t < s) red[t] += red[t + s];
        __syncthreads();
    }
    if (t == 0) out[g] = red[0] + b2[0];
}

// ---------------------------------------------------------------------------
static inline int cdiv_ll(long long a, long long b) { return (int)((a + b - 1) / b); }

extern "C" void kernel_launch(void* const* d_in, const int* in_sizes, int n_in,
                              void* d_out, int out_size, void* d_ws, size_t ws_size,
                              hipStream_t stream) {
    const float* x          = (const float*)d_in[0];
    const int*   edge_index = (const int*)  d_in[1];
    const int*   node_types = (const int*)  d_in[2];
    const int*   batch      = (const int*)  d_in[3];
    const float* emb_W      = (const float*)d_in[4];
    const float* emb_b      = (const float*)d_in[5];
    const float* ntype_emb  = (const float*)d_in[6];
    const float* gat_W      = (const float*)d_in[7];
    const float* att_src    = (const float*)d_in[8];
    const float* att_dst    = (const float*)d_in[9];
    const float* gat_b      = (const float*)d_in[10];
    const float* ln_g       = (const float*)d_in[11];
    const float* ln_b       = (const float*)d_in[12];
    const float* ga_W1      = (const float*)d_in[13];
    const float* ga_b1      = (const float*)d_in[14];
    const float* ga_W2      = (const float*)d_in[15];
    const float* ga_b2      = (const float*)d_in[16];
    const float* cls_W1     = (const float*)d_in[17];
    const float* cls_b1     = (const float*)d_in[18];
    const float* cls_W2     = (const float*)d_in[19];
    const float* cls_b2     = (const float*)d_in[20];
    float* out = (float*)d_out;

    const int N = in_sizes[0] / 64;      // 50000
    const int E = in_sizes[1] / 2;       // 600000
    const int Etot = E + N;
    const int G = NGRAPH;

    // workspace layout
    float* buf = (float*)d_ws;
    size_t off = 0;
    const size_t nh = (size_t)N * HID;
    float* h      = buf + off; off += nh;
    float* hp     = buf + off; off += nh;
    float* agg    = buf + off; off += nh;
    float* as_    = buf + off; off += (size_t)N * HEADS;
    float* ad_    = buf + off; off += (size_t)N * HEADS;
    float* smax   = buf + off; off += (size_t)N * HEADS;
    float* ssum   = buf + off; off += (size_t)N * HEADS;
    float* scores = buf + off; off += (size_t)N;
    float* wsum   = buf + off; off += HID;
    float* sums   = buf + off; off += (size_t)G * HID;
    float* cnts   = buf + off; off += G;
    float* scal   = buf + off; off += 4;     // [0]=max score, [1]=sum of exp
    _Float16* embWt = (_Float16*)(buf + off);
    _Float16* gatWt = embWt + HID * 64;
    _Float16* gaW1t = gatWt + LAYERS * HID * HID;

    const int TB = 256;

    // 0) prep f16 transposed weights
    prep_weights_kernel<<<cdiv_ll(LAYERS * HID * HID, TB), TB, 0, stream>>>(
        emb_W, gat_W, ga_W1, embWt, gatWt, gaW1t);

    // 1) node embedding: h = x @ emb_W + emb_b + ntype_emb[node_types]
    gemm_wmma_kernel<<<cdiv_ll(N, 128), TB, 0, stream>>>(
        x, embWt, emb_b, ntype_emb, node_types, h, N, 64);

    // 2) GAT layers
    for (int l = 0; l < LAYERS; ++l) {
        fill_kernel<<<cdiv_ll(N * HEADS, TB), TB, 0, stream>>>(smax, -1e30f, N * HEADS);
        fill_kernel<<<cdiv_ll(N * HEADS, TB), TB, 0, stream>>>(ssum, 0.f, N * HEADS);
        fill_kernel<<<cdiv_ll((long long)N * HID, TB), TB, 0, stream>>>(agg, 0.f, N * HID);

        gemm_wmma_kernel<<<cdiv_ll(N, 128), TB, 0, stream>>>(
            h, gatWt + (size_t)l * HID * HID, nullptr, nullptr, nullptr, hp, N, HID);

        alpha_kernel<<<cdiv_ll(N * HEADS, TB), TB, 0, stream>>>(
            hp, att_src + l * HEADS * CDIM, att_dst + l * HEADS * CDIM, as_, ad_, N);

        edge_max_kernel<<<cdiv_ll((long long)Etot * HEADS, TB), TB, 0, stream>>>(
            edge_index, as_, ad_, smax, E, Etot);
        edge_sum_kernel<<<cdiv_ll((long long)Etot * HEADS, TB), TB, 0, stream>>>(
            edge_index, as_, ad_, smax, ssum, E, Etot);
        edge_agg_kernel<<<cdiv_ll((long long)Etot * HID, TB), TB, 0, stream>>>(
            edge_index, as_, ad_, smax, ssum, hp, agg, E, Etot);

        ln_res_kernel<<<cdiv_ll((long long)N * 32, TB), TB, 0, stream>>>(
            agg, gat_b + l * HID, ln_g + l * HID, ln_b + l * HID, h, N);
    }

    // 3) global attention pooling
    gemm_wmma_kernel<<<cdiv_ll(N, 128), TB, 0, stream>>>(
        h, gaW1t, ga_b1, nullptr, nullptr, hp, N, HID);

    fill_kernel<<<1, TB, 0, stream>>>(scal, -1e30f, 1);
    fill_kernel<<<1, TB, 0, stream>>>(scal + 1, 0.f, 1);
    fill_kernel<<<1, TB, 0, stream>>>(wsum, 0.f, HID);
    fill_kernel<<<cdiv_ll(G * HID, TB), TB, 0, stream>>>(sums, 0.f, G * HID);
    fill_kernel<<<cdiv_ll(G, TB), TB, 0, stream>>>(cnts, 0.f, G);

    score_kernel<<<cdiv_ll((long long)N * 32, TB), TB, 0, stream>>>(
        hp, ga_W2, ga_b2, scores, scal, N);
    score_exp_kernel<<<cdiv_ll(N, TB), TB, 0, stream>>>(scores, scal, scal + 1, N);
    wsum_kernel<<<128, HID, 0, stream>>>(h, scores, wsum, N);

    // 4) mean pool per graph
    pool_kernel<<<cdiv_ll((long long)N * HID, TB), TB, 0, stream>>>(h, batch, sums, cnts, N);

    // 5) classifier head
    head_kernel<<<G, HID, 0, stream>>>(wsum, scal, sums, cnts,
                                       cls_W1, cls_b1, cls_W2, cls_b2, out);
}
